// CPI_regression_8675833938306
// MI455X (gfx1250) — compile-verified, matrix operand order checked by reading, not run on hardware
//
#include <hip/hip_runtime.h>
#include <hip/hip_bf16.h>
#include <math.h>

#define B_   32
#define N_   2048
#define E_   8192
#define F1_  78
#define F2_  156
#define F4_  312
#define H_   128
#define G4_  512        // 4*H
#define EMB_ 1280
#define T_   1500
#define K129 129
#define KMIX (K129*K129)   // 16641
#define SCALE_FP 4294967296.0   // Q32.32 fixed point for deterministic segment-sum

typedef __bf16 bf16_t;
typedef __attribute__((ext_vector_type(16))) __bf16 v16bf;
typedef __attribute__((ext_vector_type(8)))  float  v8f;

__device__ __forceinline__ float leaky_(float x){ return x > 0.f ? x : 0.01f * x; }
__device__ __forceinline__ float sig_(float x){ return 1.f / (1.f + __expf(-x)); }
__device__ __forceinline__ float e4_(const float4& v, int i){
  return i==0 ? v.x : (i==1 ? v.y : (i==2 ? v.z : v.w));
}

// ---------------------------------------------------------------------------
// 0) weight prep: fp32 -> bf16 copies of Wih/Whh, fused LSTM bias
// ---------------------------------------------------------------------------
__global__ void k_convert_weights(const float* __restrict__ Wih, const float* __restrict__ Whh,
                                  const float* __restrict__ bih, const float* __restrict__ bhh,
                                  bf16_t* __restrict__ wihb, bf16_t* __restrict__ whhb,
                                  float* __restrict__ gbias){
  int i = blockIdx.x * 256 + threadIdx.x;
  if (i < G4_ * EMB_) wihb[i] = (bf16_t)Wih[i];
  if (i < G4_ * H_)   whhb[i] = (bf16_t)Whh[i];
  if (i < G4_)        gbias[i] = bih[i] + bhh[i];
}

// ---------------------------------------------------------------------------
// 1) gates_x = data_pro @ Wih^T + (bih+bhh), written [T][B][512]
//    WMMA bf16 16x16x32. 256-thread WG = 8 waves; one 16-row M-tile per WG;
//    each wave owns 4 N-tiles (N=512 covered by the WG).
//    Software pipeline: A k-chunk register-prefetched while WMMAs run on the
//    LDS-staged current chunk; all 4 B fragments preloaded per k-step so the
//    4 WMMAs issue back-to-back behind a single loadcnt wait.
// ---------------------------------------------------------------------------
#define KC 128
__global__ __launch_bounds__(256)
void k_gates_gemm(const float* __restrict__ X,     // [B*T][EMB], m = b*T + t
                  const bf16_t* __restrict__ Wb,   // [512][1280] bf16
                  const float* __restrict__ gbias, // [512]
                  float* __restrict__ gates)       // [T][B][512]
{
  __shared__ float As[16][KC + 4];                 // row stride 132 f = 528 B (16B aligned)
  const int m0   = blockIdx.x * 16;
  const int tid  = threadIdx.x;
  const int wave = tid >> 5;
  const int lane = tid & 31;
  const int mloc = lane & 15;
  const int kb   = (lane >> 4) * 8;

  // A-staging assignment: 512 float4 per chunk, 2 per thread
  const int v0 = tid, v1 = tid + 256;
  const int r0 = v0 >> 5, c0 = v0 & 31;            // KC/4 = 32 float4 per row
  const int r1 = v1 >> 5, c1 = v1 & 31;
  const float* rowp0 = X + (size_t)(m0 + r0) * EMB_;
  const float* rowp1 = X + (size_t)(m0 + r1) * EMB_;

  v8f acc[4] = {};

  // prefetch chunk 0 into registers
  float4 pre0 = ((const float4*)(rowp0))[c0];
  float4 pre1 = ((const float4*)(rowp1))[c1];

  for (int kc = 0; kc < EMB_; kc += KC) {
    // commit staged registers to LDS
    *(float4*)&As[r0][c0 * 4] = pre0;
    *(float4*)&As[r1][c1 * 4] = pre1;
    __syncthreads();

    // issue next chunk's global loads; latency hides behind the WMMAs below
    if (kc + KC < EMB_) {
      pre0 = ((const float4*)(rowp0 + kc + KC))[c0];
      pre1 = ((const float4*)(rowp1 + kc + KC))[c1];
    }

    #pragma unroll
    for (int ks = 0; ks < KC/32; ++ks) {
      // build bf16 A fragment from LDS (ds_load_b128 + v_cvt_pk_bf16_f32)
      v16bf a;
      #pragma unroll
      for (int e = 0; e < 8; ++e) {
        a[e]     = (bf16_t)As[mloc][ks*32 + kb + e];
        a[8 + e] = (bf16_t)As[mloc][ks*32 + kb + 16 + e];
      }
      // preload ALL four B fragments (one clause of 8 b128 loads, one wait)
      v16bf bfr[4];
      #pragma unroll
      for (int nt = 0; nt < 4; ++nt) {
        const int n = (wave*4 + nt)*16 + (lane & 15);
        const bf16_t* bp = Wb + (size_t)n * EMB_ + kc + ks*32 + kb;
        #pragma unroll
        for (int e = 0; e < 8; ++e) { bfr[nt][e] = bp[e]; bfr[nt][8+e] = bp[16+e]; }
      }
      // back-to-back WMMAs on independent accumulators
      #pragma unroll
      for (int nt = 0; nt < 4; ++nt)
        acc[nt] = __builtin_amdgcn_wmma_f32_16x16x32_bf16(
            false, a, false, bfr[nt], (short)0, acc[nt], false, false);
    }
    __syncthreads();
  }

  #pragma unroll
  for (int nt = 0; nt < 4; ++nt) {
    const int n  = (wave*4 + nt)*16 + (lane & 15);
    const float bv = gbias[n];
    #pragma unroll
    for (int r = 0; r < 8; ++r) {
      const int m  = m0 + r + (lane >> 4) * 8;
      const int bb = m / T_;
      const int tt = m - bb * T_;
      gates[((size_t)tt * B_ + bb) * G4_ + n] = acc[nt][r] + bv;
    }
  }
}

// ---------------------------------------------------------------------------
// 2) recurrent LSTM scan: one 1024-thread WG (32 waves). Per step:
//    hw = h @ Whh^T via WMMA (wave w -> N-tile w, both M-tiles, K=128),
//    Whh B-fragments held in registers across all 1500 steps.
//    Then per-thread (b, 4 j's) gate update; c/h/sum live in registers.
//    dyn LDS: hw f32[32][512] (64KB) + hbf bf16[32][128] (8KB)
// ---------------------------------------------------------------------------
__global__ __launch_bounds__(1024)
void k_lstm_scan(const float* __restrict__ gates,   // [T][B][512]
                 const bf16_t* __restrict__ WhhB,   // [512][128] bf16
                 const int* __restrict__ lens,      // [B]
                 float* __restrict__ prot)          // [B][129]
{
  extern __shared__ char smem[];
  float  (*hw)[G4_] = (float  (*)[G4_])smem;
  bf16_t (*hbf)[H_] = (bf16_t (*)[H_])(smem + (size_t)B_ * G4_ * 4);

  const int tid  = threadIdx.x;
  const int wave = tid >> 5;        // 0..31 -> N-tile index
  const int lane = tid & 31;
  const int mloc = lane & 15;
  const int kb   = (lane >> 4) * 8;
  // update-phase ownership: b = wave, 4 consecutive j's per lane
  const int ub = wave;
  const int uj = lane * 4;
  const int mylen = lens[ub];

  float hreg[4] = {0.f,0.f,0.f,0.f};
  float creg[4] = {0.f,0.f,0.f,0.f};
  float sreg[4] = {0.f,0.f,0.f,0.f};

  for (int i = tid; i < B_ * H_; i += 1024) ((bf16_t*)hbf)[i] = (bf16_t)0.0f;

  // preload Whh B-fragments (persist in VGPRs across the whole scan)
  v16bf bfrag[4];
  #pragma unroll
  for (int ks = 0; ks < 4; ++ks) {
    const bf16_t* bp = WhhB + (size_t)(wave*16 + (lane & 15)) * H_ + ks*32 + kb;
    #pragma unroll
    for (int e = 0; e < 8; ++e) { bfrag[ks][e] = bp[e]; bfrag[ks][8+e] = bp[16+e]; }
  }
  __syncthreads();

  for (int t = 0; t < T_; ++t) {
    // ---- GEMM phase: hw[0:32][16w:16w+16] = h @ Whh^T
    v8f acc0 = {}, acc1 = {};
    #pragma unroll
    for (int ks = 0; ks < 4; ++ks) {
      v16bf a0, a1;
      #pragma unroll
      for (int e = 0; e < 8; ++e) {
        a0[e]   = hbf[mloc     ][ks*32 + kb + e];
        a0[8+e] = hbf[mloc     ][ks*32 + kb + 16 + e];
        a1[e]   = hbf[mloc + 16][ks*32 + kb + e];
        a1[8+e] = hbf[mloc + 16][ks*32 + kb + 16 + e];
      }
      acc0 = __builtin_amdgcn_wmma_f32_16x16x32_bf16(false, a0, false, bfrag[ks], (short)0, acc0, false, false);
      acc1 = __builtin_amdgcn_wmma_f32_16x16x32_bf16(false, a1, false, bfrag[ks], (short)0, acc1, false, false);
    }
    const int n = wave*16 + (lane & 15);
    #pragma unroll
    for (int r = 0; r < 8; ++r) {
      const int m = r + (lane >> 4) * 8;
      hw[m     ][n] = acc0[r];
      hw[m + 16][n] = acc1[r];
    }
    __syncthreads();

    // ---- update phase: thread owns (ub, uj..uj+3)
    const float* gp  = gates + ((size_t)t * B_ + ub) * G4_;
    const float* hp  = &hw[ub][0];
    float4 xi = *(const float4*)(gp + 0*H_ + uj);
    float4 xf = *(const float4*)(gp + 1*H_ + uj);
    float4 xg = *(const float4*)(gp + 2*H_ + uj);
    float4 xo = *(const float4*)(gp + 3*H_ + uj);
    float4 hi = *(const float4*)(hp + 0*H_ + uj);
    float4 hf = *(const float4*)(hp + 1*H_ + uj);
    float4 hg = *(const float4*)(hp + 2*H_ + uj);
    float4 ho = *(const float4*)(hp + 3*H_ + uj);
    const bool mk = (t < mylen);
    #pragma unroll
    for (int e = 0; e < 4; ++e) {
      float iv = sig_ (e4_(xi,e) + e4_(hi,e));
      float fv = sig_ (e4_(xf,e) + e4_(hf,e));
      float gv = tanhf(e4_(xg,e) + e4_(hg,e));
      float ov = sig_ (e4_(xo,e) + e4_(ho,e));
      float cn = fv * creg[e] + iv * gv;
      float hn = ov * tanhf(cn);
      if (mk) { creg[e] = cn; hreg[e] = hn; sreg[e] += hn; }
      hbf[ub][uj + e] = (bf16_t)hreg[e];
    }
    __syncthreads();
  }

  const float inv_t = 1.0f / (float)T_;
  #pragma unroll
  for (int e = 0; e < 4; ++e) prot[ub * K129 + uj + e] = sreg[e] * inv_t;
  if (lane == 0) prot[ub * K129 + H_] = 1.0f;
}

// ---------------------------------------------------------------------------
// 3) molecule branch (small, deterministic)
// ---------------------------------------------------------------------------
__global__ void k_zero32(unsigned int* p, int n){
  int i = blockIdx.x * 256 + threadIdx.x;
  if (i < n) p[i] = 0u;
}

__global__ void k_deg(const int* __restrict__ ei, float* __restrict__ cnt){
  int e = blockIdx.x * 256 + threadIdx.x;
  if (e < E_) atomicAdd(&cnt[ei[E_ + e]], 1.0f);   // exact small ints -> deterministic
}

// deterministic segment-sum via Q32.32 integer atomics (associative)
__global__ void k_scatter(const int* __restrict__ ei, const float* __restrict__ x,
                          unsigned long long* __restrict__ msg, int d){
  int i = blockIdx.x * 256 + threadIdx.x;
  if (i >= E_ * d) return;
  int e = i / d, f = i - e * d;
  int s  = ei[e];
  int dn = ei[E_ + e];
  long long q = (long long)((double)x[(size_t)s * d + f] * SCALE_FP);
  atomicAdd(&msg[(size_t)dn * d + f], (unsigned long long)q);
}

__global__ void k_sage_lin(const float* __restrict__ x,
                           const unsigned long long* __restrict__ msg,
                           const float* __restrict__ cnt,
                           const float* __restrict__ Wl, const float* __restrict__ bl,
                           const float* __restrict__ Wr,
                           float* __restrict__ out, int din, int dout){
  int i = blockIdx.x * 256 + threadIdx.x;
  if (i >= N_ * dout) return;
  int nn = i / dout, o = i - nn * dout;
  float invc = 1.0f / fmaxf(cnt[nn], 1.0f);
  const float* xr = x + (size_t)nn * din;
  const unsigned long long* mr = msg + (size_t)nn * din;
  const float* wl = Wl + (size_t)o * din;
  const float* wr = Wr + (size_t)o * din;
  float acc = bl[o];
  for (int k = 0; k < din; ++k) {
    float aggk = (float)((double)(long long)mr[k] * (1.0 / SCALE_FP)) * invc;
    acc += aggk * wl[k] + xr[k] * wr[k];
  }
  out[i] = leaky_(acc);
}

__global__ void k_pool_max(const float* __restrict__ x3, const int* __restrict__ mb,
                           float* __restrict__ pooled){
  int i = blockIdx.x * 256 + threadIdx.x;
  if (i >= B_ * F4_) return;
  int b = i / F4_, f = i - b * F4_;
  float m = -3.4e38f;
  for (int n = 0; n < N_; ++n)
    if (mb[n] == b) m = fmaxf(m, x3[(size_t)n * F4_ + f]);
  pooled[i] = m;
}

__global__ void k_fcg1(const float* __restrict__ pooled, const float* __restrict__ W,
                       const float* __restrict__ bias, float* __restrict__ comp){
  int i = blockIdx.x * 256 + threadIdx.x;
  if (i >= B_ * H_) return;
  int b = i / H_, o = i - b * H_;
  float acc = bias[o];
  for (int k = 0; k < F4_; ++k) acc += pooled[b * F4_ + k] * W[(size_t)o * F4_ + k];
  comp[b * K129 + o] = acc;
  if (o == 0) comp[b * K129 + H_] = 1.0f;
}

// ---------------------------------------------------------------------------
// 4) fusion head
// ---------------------------------------------------------------------------
__global__ void k_outer(const float* __restrict__ prot, const float* __restrict__ comp,
                        float* __restrict__ outT){   // [KMIX][B] transposed
  int idx = blockIdx.x * 256 + threadIdx.x;
  if (idx >= KMIX * B_) return;
  int k = idx / B_, b = idx - k * B_;
  int i = k / K129, j = k - i * K129;
  outT[(size_t)k * B_ + b] = prot[b * K129 + i] * comp[b * K129 + j];
}

__global__ __launch_bounds__(256)
void k_mix(const float* __restrict__ outT, const float* __restrict__ W,
           const float* __restrict__ bias, float* __restrict__ mixo){ // [B][1024]
  const int r = blockIdx.x;           // output row 0..1023
  const int g = threadIdx.x >> 5;     // k-group 0..7
  const int b = threadIdx.x & 31;     // batch lane (coalesced axis of outT)
  const float* wr = W + (size_t)r * KMIX;
  float acc = 0.f;
  for (int k = g; k < KMIX; k += 8)
    acc += wr[k] * outT[(size_t)k * B_ + b];
  __shared__ float red[8][32];
  red[g][b] = acc;
  __syncthreads();
  if (g == 0) {
    float s = bias[r];
    #pragma unroll
    for (int gg = 0; gg < 8; ++gg) s += red[gg][b];
    mixo[(size_t)b * 1024 + r] = leaky_(s);
  }
}

__global__ void k_fc1(const float* __restrict__ mixo, const float* __restrict__ prot,
                      const float* __restrict__ comp, const float* __restrict__ W,
                      const float* __restrict__ bias, float* __restrict__ out){ // [B][512]
  int i = blockIdx.x * 256 + threadIdx.x;
  if (i >= B_ * 512) return;
  int b = i / 512, o = i - b * 512;
  const float* wr = W + (size_t)o * (1024 + 2 * K129);
  float acc = bias[o];
  for (int k = 0; k < 1024; ++k) acc += mixo[(size_t)b * 1024 + k] * wr[k];
  for (int k = 0; k < K129; ++k) acc += prot[b * K129 + k] * wr[1024 + k];
  for (int k = 0; k < K129; ++k) acc += comp[b * K129 + k] * wr[1024 + K129 + k];
  out[i] = leaky_(acc);
}

__global__ void k_fc2(const float* __restrict__ f1, const float* __restrict__ W,
                      const float* __restrict__ bias, float* __restrict__ out){
  int b = blockIdx.x * 64 + threadIdx.x;
  if (b >= B_) return;
  float acc = bias[0];
  for (int k = 0; k < 512; ++k) acc += f1[(size_t)b * 512 + k] * W[k];
  out[b] = acc;
}

// ---------------------------------------------------------------------------
extern "C" void kernel_launch(void* const* d_in, const int* in_sizes, int n_in,
                              void* d_out, int out_size, void* d_ws, size_t ws_size,
                              hipStream_t stream) {
  (void)in_sizes; (void)n_in; (void)out_size; (void)ws_size;

  const float* mol_x   = (const float*)d_in[0];
  const float* dpro    = (const float*)d_in[1];
  const int*   ei      = (const int*)  d_in[2];
  const int*   mbatch  = (const int*)  d_in[3];
  const int*   plen    = (const int*)  d_in[4];
  const float* s1Wl = (const float*)d_in[5];  const float* s1bl = (const float*)d_in[6];
  const float* s1Wr = (const float*)d_in[7];
  const float* s2Wl = (const float*)d_in[8];  const float* s2bl = (const float*)d_in[9];
  const float* s2Wr = (const float*)d_in[10];
  const float* s3Wl = (const float*)d_in[11]; const float* s3bl = (const float*)d_in[12];
  const float* s3Wr = (const float*)d_in[13];
  const float* fg1W = (const float*)d_in[14]; const float* fg1b = (const float*)d_in[15];
  const float* Wih  = (const float*)d_in[16]; const float* Whh  = (const float*)d_in[17];
  const float* bih  = (const float*)d_in[18]; const float* bhh  = (const float*)d_in[19];
  const float* mixW = (const float*)d_in[20]; const float* mixb = (const float*)d_in[21];
  const float* fc1W = (const float*)d_in[22]; const float* fc1b = (const float*)d_in[23];
  const float* fc2W = (const float*)d_in[24]; const float* fc2b = (const float*)d_in[25];
  float* out = (float*)d_out;

  // workspace carve-out
  char* p = (char*)d_ws;
  auto carve = [&](size_t bytes) { void* r = (void*)p; p += (bytes + 255) & ~(size_t)255; return r; };
  float*  gates = (float*) carve((size_t)T_ * B_ * G4_ * 4);   // 98.3 MB
  bf16_t* wihb  = (bf16_t*)carve((size_t)G4_ * EMB_ * 2);
  bf16_t* whhb  = (bf16_t*)carve((size_t)G4_ * H_ * 2);
  float*  gbias = (float*) carve(G4_ * 4);
  float*  prot  = (float*) carve(B_ * K129 * 4);
  float*  comp  = (float*) carve(B_ * K129 * 4);
  float*  cnt   = (float*) carve(N_ * 4);
  unsigned long long* msg = (unsigned long long*)carve((size_t)N_ * F2_ * 8);
  float*  x1    = (float*) carve((size_t)N_ * F2_ * 4);
  float*  x2    = (float*) carve((size_t)N_ * F2_ * 4);
  float*  x3    = (float*) carve((size_t)N_ * F4_ * 4);
  float*  pooled= (float*) carve(B_ * F4_ * 4);
  float*  outT  = (float*) carve((size_t)KMIX * B_ * 4);
  float*  mixo  = (float*) carve(B_ * 1024 * 4);
  float*  f1o   = (float*) carve(B_ * 512 * 4);

  // --- protein branch ---
  k_convert_weights<<<(G4_ * EMB_ + 255) / 256, 256, 0, stream>>>(Wih, Whh, bih, bhh, wihb, whhb, gbias);
  k_gates_gemm<<<(B_ * T_) / 16, 256, 0, stream>>>(dpro, wihb, gbias, gates);
  const size_t scan_lds = (size_t)B_ * G4_ * 4 + (size_t)B_ * H_ * 2;  // 72 KB
  k_lstm_scan<<<1, 1024, scan_lds, stream>>>(gates, whhb, plen, prot);

  // --- molecule branch ---
  k_zero32<<<(N_ + 255) / 256, 256, 0, stream>>>((unsigned int*)cnt, N_);
  k_deg<<<E_ / 256, 256, 0, stream>>>(ei, cnt);

  // layer 1: 78 -> 156
  k_zero32<<<((N_ * F1_ * 2) + 255) / 256, 256, 0, stream>>>((unsigned int*)msg, N_ * F1_ * 2);
  k_scatter<<<(E_ * F1_ + 255) / 256, 256, 0, stream>>>(ei, mol_x, msg, F1_);
  k_sage_lin<<<(N_ * F2_ + 255) / 256, 256, 0, stream>>>(mol_x, msg, cnt, s1Wl, s1bl, s1Wr, x1, F1_, F2_);
  // layer 2: 156 -> 156
  k_zero32<<<((N_ * F2_ * 2) + 255) / 256, 256, 0, stream>>>((unsigned int*)msg, N_ * F2_ * 2);
  k_scatter<<<(E_ * F2_ + 255) / 256, 256, 0, stream>>>(ei, x1, msg, F2_);
  k_sage_lin<<<(N_ * F2_ + 255) / 256, 256, 0, stream>>>(x1, msg, cnt, s2Wl, s2bl, s2Wr, x2, F2_, F2_);
  // layer 3: 156 -> 312
  k_zero32<<<((N_ * F2_ * 2) + 255) / 256, 256, 0, stream>>>((unsigned int*)msg, N_ * F2_ * 2);
  k_scatter<<<(E_ * F2_ + 255) / 256, 256, 0, stream>>>(ei, x2, msg, F2_);
  k_sage_lin<<<(N_ * F4_ + 255) / 256, 256, 0, stream>>>(x2, msg, cnt, s3Wl, s3bl, s3Wr, x3, F2_, F4_);

  k_pool_max<<<(B_ * F4_ + 255) / 256, 256, 0, stream>>>(x3, mbatch, pooled);
  k_fcg1<<<(B_ * H_ + 255) / 256, 256, 0, stream>>>(pooled, fg1W, fg1b, comp);

  // --- fusion head ---
  k_outer<<<((size_t)KMIX * B_ + 255) / 256, 256, 0, stream>>>(prot, comp, outT);
  k_mix<<<1024, 256, 0, stream>>>(outT, mixW, mixb, mixo);
  k_fc1<<<(B_ * 512 + 255) / 256, 256, 0, stream>>>(mixo, prot, comp, fc1W, fc1b, f1o);
  k_fc2<<<1, 64, 0, stream>>>(f1o, fc2W, fc2b, out);
}
